// SFC_GAT_59622736003361
// MI455X (gfx1250) — matrix operand order repaired and checked by reading, not compiled, not run on hardware
//
#include <hip/hip_runtime.h>
#include <hip/hip_bf16.h>

typedef float  v8f  __attribute__((ext_vector_type(8)));
typedef float  v4f  __attribute__((ext_vector_type(4)));
typedef float  v2f  __attribute__((ext_vector_type(2)));
typedef __bf16 v16bf __attribute__((ext_vector_type(16)));
typedef __bf16 v8bf  __attribute__((ext_vector_type(8)));
typedef __bf16 v4bf  __attribute__((ext_vector_type(4)));

#define N_NODES 3072
#define IN_FEAT 64
#define OUT_FEAT 16
#define H_HEADS 8
#define LRELU_ALPHA 0.2f
#define NEG_INF_V -9.0e15f

// ---------------------------------------------------------------------------
// Build dense adjacency as a bitmask (1 bit per (src,dst) pair).
// ---------------------------------------------------------------------------
__global__ void k_adj(const long long* __restrict__ ei, unsigned* __restrict__ adj, int E) {
    int j = blockIdx.x * blockDim.x + threadIdx.x;
    if (j >= E) return;
    int s = (int)ei[j];
    int d = (int)ei[E + j];
    long long bit = (long long)s * N_NODES + d;
    atomicOr(&adj[bit >> 5], 1u << (bit & 31));
}

// ---------------------------------------------------------------------------
// Wh[h,n,o] = sum_f h[n,f] * W[h,f,o]  via V_WMMA_F32_16X16X4_F32.
// One wave per (16-row block, head). Emits:
//   Whf  [h][n][o] f32  (for e1/e2)
//   WhT  [h][o][n] bf16 (B-fragment friendly: K(=n) contiguous per column o)
// C-fragment: vgpr k = rows (k, k+8) per lane half, N = lane&15 -> the 8
// accumulator values per lane are M-consecutive => one 16B store to WhT.
// ---------------------------------------------------------------------------
__global__ void k_wh(const float* __restrict__ hsrc, const float* __restrict__ W,
                     float* __restrict__ Whf, __bf16* __restrict__ WhT) {
    const int lane = threadIdx.x;
    const int hi = lane >> 4;        // lane half selects K pair
    const int Ml = lane & 15;        // row (A) / col (B)
    const int rb = blockIdx.x, head = blockIdx.y;
    const int row = rb * 16 + Ml;
    const float* Wp = W + (size_t)head * IN_FEAT * OUT_FEAT;

    v8f c = {};
    for (int kb = 0; kb < IN_FEAT; kb += 4) {
        const int k = kb + 2 * hi;
        v2f a, b;
        a[0] = hsrc[(size_t)row * IN_FEAT + k];
        a[1] = hsrc[(size_t)row * IN_FEAT + k + 1];
        b[0] = Wp[(size_t)k * OUT_FEAT + Ml];
        b[1] = Wp[(size_t)(k + 1) * OUT_FEAT + Ml];
        c = __builtin_amdgcn_wmma_f32_16x16x4_f32(false, a, false, b, (short)0, c,
                                                  false, false);
    }
    v8bf cb;
#pragma unroll
    for (int k = 0; k < 8; ++k) {
        const int M = k + 8 * hi;
        Whf[(size_t)head * N_NODES * OUT_FEAT + (size_t)(rb * 16 + M) * OUT_FEAT + Ml]
            = c[k];
        cb[k] = (__bf16)c[k];
    }
    // WhT[head][Ml][rb*16 + 8*hi + 0..7] : contiguous, 16B aligned
    *(v8bf*)&WhT[(size_t)head * OUT_FEAT * N_NODES + (size_t)Ml * N_NODES +
                 rb * 16 + 8 * hi] = cb;
}

// ---------------------------------------------------------------------------
// e1[h,n] = Wh[h,n,:]·a1[h]; e2[h,n] = Wh[h,n,:]·a2[h]
// ---------------------------------------------------------------------------
__global__ void k_e(const float* __restrict__ Whf, const float* __restrict__ a,
                    float* __restrict__ e1, float* __restrict__ e2) {
    int idx = blockIdx.x * blockDim.x + threadIdx.x;
    if (idx >= H_HEADS * N_NODES) return;
    int head = idx / N_NODES;
    int n = idx - head * N_NODES;
    const float* w  = Whf + (size_t)head * N_NODES * OUT_FEAT + (size_t)n * OUT_FEAT;
    const float* a1 = a + (size_t)head * 2 * OUT_FEAT;
    const float* a2 = a1 + OUT_FEAT;
    float s1 = 0.f, s2 = 0.f;
#pragma unroll
    for (int o = 0; o < OUT_FEAT; ++o) { s1 += w[o] * a1[o]; s2 += w[o] * a2[o]; }
    e1[idx] = s1;
    e2[idx] = s2;
}

// ---------------------------------------------------------------------------
// Fused per-(head, 16-row-block) attention. 256 threads = 8 wave32.
// Phase 1: vectorized (4 m / thread) logits -> LDS bf16 tile.
// Phase 2: row-wise softmax (vectorized b64 LDS traffic).
// Phase 3: P(16x3072) x Wh(3072x16) via V_WMMA_F32_16X16X32_BF16,
//          A from 2x ds_load_b128, B from 2x 16B vector loads of WhT.
// Phase 4: cross-wave reduce, 1/rowsum, ELU, store.
// ---------------------------------------------------------------------------
__global__ void k_gat(const float* __restrict__ S, const unsigned* __restrict__ adj,
                      const __bf16* __restrict__ WhT,
                      const float* __restrict__ e1g, const float* __restrict__ e2g,
                      const float* __restrict__ rg, float* __restrict__ out) {
    extern __shared__ char smemRaw[];
    __bf16* sP   = (__bf16*)smemRaw;                             // 16*3072 bf16 (96 KB)
    float*  sMax = (float*)(smemRaw + (size_t)16 * N_NODES * 2); // 16
    float*  sSum = sMax + 16;                                    // 16
    float*  sE1  = sSum + 16;                                    // 16
    float*  sPart = sE1 + 16;                                    // 8*256 f32 (8 KB)

    const int tid = threadIdx.x;
    const int rb = blockIdx.x, head = blockIdx.y;
    const int wv = tid >> 5, lane = tid & 31;
    const float gamma = log1pf(expf(rg[head]));                  // softplus

    if (tid < 16) sE1[tid] = e1g[head * N_NODES + rb * 16 + tid];
    __syncthreads();

    // ---- Phase 1: masked/scaled logits into LDS, 4 columns per thread ----
    for (int idx4 = tid; idx4 < 16 * (N_NODES / 4); idx4 += 256) {
        const int r  = idx4 / (N_NODES / 4);
        const int m0 = (idx4 - r * (N_NODES / 4)) * 4;
        const int row = rb * 16 + r;
        const v4f Sv  = *(const v4f*)&S[(size_t)row * N_NODES + m0];
        const v4f e2v = *(const v4f*)&e2g[head * N_NODES + m0];
        const long long base = (long long)row * N_NODES + m0;    // mult of 4
        const unsigned bits4 = adj[base >> 5] >> (base & 31);
        const float e1r = sE1[r];
        v4bf pv;
#pragma unroll
        for (int j = 0; j < 4; ++j) {
            float x = e1r + e2v[j];
            x = x > 0.f ? x : LRELU_ALPHA * x;                   // leaky relu
            float v = ((bits4 >> j) & 1u) ? x : NEG_INF_V;       // mask BEFORE scale
            v *= (1.f + gamma * Sv[j]);                          // reference order
            pv[j] = (__bf16)v;
        }
        *(v4bf*)&sP[r * N_NODES + m0] = pv;                      // 8B aligned
    }
    __syncthreads();

    // ---- Phase 2a: per-row max (wave wv owns rows 2wv, 2wv+1) ------------
    for (int rr = 0; rr < 2; ++rr) {
        const int r = wv * 2 + rr;
        float mx = -3.0e38f;
        for (int mb = lane; mb < N_NODES / 4; mb += 32) {
            const v4bf v = *(const v4bf*)&sP[r * N_NODES + mb * 4];
#pragma unroll
            for (int j = 0; j < 4; ++j) mx = fmaxf(mx, (float)v[j]);
        }
        for (int off = 16; off > 0; off >>= 1)
            mx = fmaxf(mx, __shfl_xor(mx, off, 32));
        if (lane == 0) sMax[r] = mx;
    }
    __syncthreads();

    // ---- Phase 2b: p = exp(v - max) in-place, row sums -------------------
    for (int rr = 0; rr < 2; ++rr) {
        const int r = wv * 2 + rr;
        const float mxr = sMax[r];
        float sm = 0.f;
        for (int mb = lane; mb < N_NODES / 4; mb += 32) {
            v4bf v = *(v4bf*)&sP[r * N_NODES + mb * 4];
            v4bf p;
#pragma unroll
            for (int j = 0; j < 4; ++j) {
                float pf = expf((float)v[j] - mxr);
                p[j] = (__bf16)pf;
                sm += pf;
            }
            *(v4bf*)&sP[r * N_NODES + mb * 4] = p;
        }
        for (int off = 16; off > 0; off >>= 1) sm += __shfl_xor(sm, off, 32);
        if (lane == 0) sSum[r] = sm;
    }
    __syncthreads();

    // ---- Phase 3: O = P x Wh via bf16 WMMA, K chunks split across waves --
    v8f acc = {};
    const __bf16* WhTb = WhT + (size_t)head * OUT_FEAT * N_NODES;
    const int Ml = lane & 15, hi = lane >> 4;
    for (int ch = wv; ch < N_NODES / 32; ch += 8) {
        const int k0 = ch * 32;
        // A (16x32 bf16): lane = row Ml, K = hi*8 + {0..7} and hi*8+16 + {0..7}
        const v8bf alo = *(const v8bf*)&sP[Ml * N_NODES + k0 + hi * 8];        // 16B
        const v8bf ahi = *(const v8bf*)&sP[Ml * N_NODES + k0 + hi * 8 + 16];   // 16B
        const v16bf A = __builtin_shufflevector(alo, ahi,
            0, 1, 2, 3, 4, 5, 6, 7, 8, 9, 10, 11, 12, 13, 14, 15);
        // B (32x16 bf16): lane = col Ml, K = hi*8 + {0..7} and hi*8+16 + {0..7}
        const v8bf blo = *(const v8bf*)&WhTb[(size_t)Ml * N_NODES + k0 + hi * 8];
        const v8bf bhi = *(const v8bf*)&WhTb[(size_t)Ml * N_NODES + k0 + hi * 8 + 16];
        const v16bf B = __builtin_shufflevector(blo, bhi,
            0, 1, 2, 3, 4, 5, 6, 7, 8, 9, 10, 11, 12, 13, 14, 15);
        acc = __builtin_amdgcn_wmma_f32_16x16x32_bf16(false, A, false, B, (short)0,
                                                      acc, false, false);
    }
#pragma unroll
    for (int k = 0; k < 8; ++k) sPart[wv * 256 + k * 32 + lane] = acc[k];
    __syncthreads();

    // ---- Phase 4: cross-wave reduce, normalize, ELU, store ---------------
    {
        const int k = tid >> 5, ln = tid & 31;
        float s = 0.f;
#pragma unroll
        for (int w = 0; w < 8; ++w) s += sPart[w * 256 + k * 32 + ln];
        const int rloc = k + 8 * (ln >> 4);
        const int col = ln & 15;
        float val = s / sSum[rloc];
        val = val > 0.f ? val : (expf(val) - 1.f);               // ELU (alpha=1)
        out[(size_t)(rb * 16 + rloc) * (H_HEADS * OUT_FEAT) + head * OUT_FEAT + col] = val;
    }
}

// ---------------------------------------------------------------------------
extern "C" void kernel_launch(void* const* d_in, const int* in_sizes, int n_in,
                              void* d_out, int out_size, void* d_ws, size_t ws_size,
                              hipStream_t stream) {
    (void)n_in; (void)out_size; (void)ws_size;
    const float*     h  = (const float*)d_in[0];
    const long long* ei = (const long long*)d_in[1];   // reference dtype int64
    const float*     S  = (const float*)d_in[2];
    const float*     W  = (const float*)d_in[3];
    const float*     a  = (const float*)d_in[4];
    const float*     rg = (const float*)d_in[5];
    float* out = (float*)d_out;
    const int E = in_sizes[1] / 2;

    char* ws = (char*)d_ws;
    const size_t ADJ_BYTES = (size_t)N_NODES * N_NODES / 8;               // 1.18 MB
    const size_t WHF_BYTES = (size_t)H_HEADS * N_NODES * OUT_FEAT * 4;    // 1.57 MB
    const size_t WHT_BYTES = WHF_BYTES / 2;                               // 0.79 MB
    unsigned* adj = (unsigned*)ws;
    float*    Whf = (float*)(ws + ADJ_BYTES);
    __bf16*   WhT = (__bf16*)(ws + ADJ_BYTES + WHF_BYTES);
    float*    e1  = (float*)(ws + ADJ_BYTES + WHF_BYTES + WHT_BYTES);
    float*    e2  = e1 + H_HEADS * N_NODES;

    hipMemsetAsync(adj, 0, ADJ_BYTES, stream);
    k_adj<<<(E + 255) / 256, 256, 0, stream>>>(ei, adj, E);
    k_wh<<<dim3(N_NODES / 16, H_HEADS), 32, 0, stream>>>(h, W, Whf, WhT);
    k_e<<<(H_HEADS * N_NODES + 255) / 256, 256, 0, stream>>>(Whf, a, e1, e2);

    const size_t shmem = (size_t)16 * N_NODES * 2     // bf16 P tile
                       + 4 * (16 + 16 + 16)           // max / sum / e1
                       + 4 * 8 * 256;                 // cross-wave partials
    k_gat<<<dim3(N_NODES / 16, H_HEADS), 256, shmem, stream>>>(S, adj, WhT, e1, e2,
                                                               rg, out);
}
